// EGTransformer_71511205478631
// MI455X (gfx1250) — compile-verified, matrix operand order checked by reading, not run on hardware
//
#include <hip/hip_runtime.h>
#include <hip/hip_bf16.h>

typedef _Float16 h16;
typedef __attribute__((ext_vector_type(16))) _Float16 v16h;
typedef __attribute__((ext_vector_type(8)))  _Float16 v8h;
typedef __attribute__((ext_vector_type(8)))  float    v8f;

#define Bk 4
#define Hk 64
#define Wk 64
#define Ck 512
#define NHk 16
#define HDk 32
#define Lk (Hk*Wk)          // 4096
#define Mrows (Bk*Lk)       // 16384
#define Sk 64
#define NWk 256

// ---------------------------------------------------------------- helpers
__device__ __forceinline__ v16h hcombine(v8h a, v8h b) {
  return __builtin_shufflevector(a, b, 0,1,2,3,4,5,6,7,8,9,10,11,12,13,14,15);
}

// Load a 16x32 f16 WMMA A/B fragment. Rows (M or N index) advance by ldh
// halves; K is contiguous. Per ISA layout: lanes 0-15 hold K {0..7,16..23},
// lanes 16-31 hold K {8..15,24..31}, row = lane%16.
__device__ __forceinline__ v16h load_frag(const h16* __restrict__ base, int ldh, int lane) {
  const int hs = (lane >> 4) & 1;
  const int lr = lane & 15;
  const h16* p = base + (size_t)lr * (size_t)ldh + hs * 8;
  v8h lo = *(const v8h*)(p);
  v8h hi = *(const v8h*)(p + 16);
  return hcombine(lo, hi);
}

__device__ __forceinline__ v8f wmma16(v16h a, v16h b, v8f c) {
  return __builtin_amdgcn_wmma_f32_16x16x32_f16(false, a, false, b, (short)0, c, false, false);
}

__device__ __forceinline__ v8f vzero8() { v8f z = {0.f,0.f,0.f,0.f,0.f,0.f,0.f,0.f}; return z; }

// ---------------------------------------------------------------- weight convert: w[K,N] f32 -> wT[N,K] f16
__global__ void wconv_kernel(const float* __restrict__ w, h16* __restrict__ wt, int K, int N) {
  int t = blockIdx.x * blockDim.x + threadIdx.x;
  if (t >= K * N) return;
  int n = t % N;
  int k = t / N;
  wt[(size_t)n * K + k] = (h16)w[(size_t)k * N + n];
}

// ---------------------------------------------------------------- LayerNorm -> f16, wave per row (C=512)
__global__ __launch_bounds__(256)
void ln_kernel(const float* __restrict__ x, const float* __restrict__ g,
               const float* __restrict__ b, h16* __restrict__ out) {
  const int lane = threadIdx.x & 31;
  const int wv   = threadIdx.x >> 5;
  const int row  = blockIdx.x * 8 + wv;
  const float* xr = x + (size_t)row * Ck;
  float v[16];
  float s = 0.f, sq = 0.f;
#pragma unroll
  for (int i = 0; i < 16; ++i) {
    float t = xr[i * 32 + lane];
    v[i] = t; s += t; sq += t * t;
  }
#pragma unroll
  for (int m = 16; m; m >>= 1) { s += __shfl_xor(s, m, 32); sq += __shfl_xor(sq, m, 32); }
  float mean = s * (1.f / 512.f);
  float var  = sq * (1.f / 512.f) - mean * mean;
  float rstd = rsqrtf(var + 1e-5f);
  h16* orow = out + (size_t)row * Ck;
#pragma unroll
  for (int i = 0; i < 16; ++i) {
    int c = i * 32 + lane;
    orow[c] = (h16)((v[i] - mean) * rstd * g[c] + b[c]);
  }
}

// ---------------------------------------------------------------- generic WMMA GEMM
// C[M,N] = A[M,K](f16) * BT[N,K](f16)   (B pre-transposed so K is contiguous)
// mode 0: store f16 (no bias)            -> outH
// mode 1: + bias + resIn, store f32      -> outF   (resIn/outF may alias)
// mode 2: + bias, exact GELU, store f16  -> outH
__global__ __launch_bounds__(256)
void gemm_kernel(const h16* __restrict__ A, const h16* __restrict__ BT,
                 const float* __restrict__ bias, const float* __restrict__ resIn,
                 float* __restrict__ outF, h16* __restrict__ outH,
                 int N, int K, int mode) {
  const int lane = threadIdx.x & 31;
  const int wv   = threadIdx.x >> 5;
  const int hs   = lane >> 4;
  const int lr   = lane & 15;
  const int m0   = blockIdx.y * 256 + wv * 32;   // wave: 32 rows x 64 cols
  const int n0   = blockIdx.x * 64;

  v8f acc[2][4];
#pragma unroll
  for (int i = 0; i < 2; ++i)
#pragma unroll
    for (int j = 0; j < 4; ++j) acc[i][j] = vzero8();

  for (int kk = 0; kk < K; kk += 32) {
    v16h a0 = load_frag(A + (size_t)m0 * K + kk, K, lane);
    v16h a1 = load_frag(A + (size_t)(m0 + 16) * K + kk, K, lane);
#pragma unroll
    for (int nt = 0; nt < 4; ++nt) {
      v16h bfr = load_frag(BT + (size_t)(n0 + nt * 16) * K + kk, K, lane);
      acc[0][nt] = wmma16(a0, bfr, acc[0][nt]);
      acc[1][nt] = wmma16(a1, bfr, acc[1][nt]);
    }
  }

#pragma unroll
  for (int mi = 0; mi < 2; ++mi)
#pragma unroll
    for (int nt = 0; nt < 4; ++nt)
#pragma unroll
      for (int r = 0; r < 8; ++r) {
        int gm = m0 + mi * 16 + r + 8 * hs;
        int gn = n0 + nt * 16 + lr;
        size_t o = (size_t)gm * N + gn;
        float val = acc[mi][nt][r];
        if (mode == 0) {
          outH[o] = (h16)val;
        } else if (mode == 1) {
          outF[o] = val + bias[gn] + resIn[o];
        } else {
          val += bias[gn];
          float gl = 0.5f * val * (1.0f + erff(val * 0.70710678118654752f));
          outH[o] = (h16)gl;
        }
      }
}

// ---------------------------------------------------------------- attention pass A: sum|scores+0.1*rpe| per (window,head)
__global__ __launch_bounds__(32)
void attn_mean_kernel(const h16* __restrict__ qkv, const float* __restrict__ rpe,
                      float* __restrict__ Mpart, int idx) {
  const int lane = threadIdx.x;
  const int hs = lane >> 4, lr = lane & 15;
  const int w = blockIdx.x >> 4;
  const int h = blockIdx.x & 15;
  int start, sst;
  if (idx == 0) { start = w * 64; sst = 1; }
  else          { start = (w >> 6) * Lk + (w & 63); sst = 64; }
  const int ldq = sst * (3 * Ck);

  const h16* qb = qkv + (size_t)start * (3 * Ck) + h * HDk;
  const h16* kb = qkv + (size_t)start * (3 * Ck) + Ck + h * HDk;
  const float* rp = rpe + h * (Sk * Sk);

  v16h kf[4];
#pragma unroll
  for (int nt = 0; nt < 4; ++nt)
    kf[nt] = load_frag(kb + (size_t)(nt * 16) * ldq, ldq, lane);

  float sum = 0.f;
#pragma unroll
  for (int mt = 0; mt < 4; ++mt) {
    v16h qa = load_frag(qb + (size_t)(mt * 16) * ldq, ldq, lane);
#pragma unroll
    for (int nt = 0; nt < 4; ++nt) {
      v8f c = vzero8();
      c = wmma16(qa, kf[nt], c);
#pragma unroll
      for (int r = 0; r < 8; ++r) {
        float val = c[r] + 0.1f * rp[(mt * 16 + r + 8 * hs) * Sk + nt * 16 + lr];
        sum += fabsf(val);
      }
    }
  }
#pragma unroll
  for (int m = 16; m; m >>= 1) sum += __shfl_xor(sum, m, 32);
  if (lane == 0) Mpart[blockIdx.x] = sum;
}

// ---------------------------------------------------------------- gating M: per-window mean, global max, clamp
__global__ __launch_bounds__(256)
void mfinal_kernel(const float* __restrict__ Mpart, float* __restrict__ Mfull) {
  __shared__ float sm[256];
  int w = threadIdx.x;
  float s = 0.f;
#pragma unroll
  for (int h = 0; h < 16; ++h) s += Mpart[w * 16 + h];
  float mean = s * (1.0f / 65536.0f);   // NH*S*S
  sm[w] = mean;
  __syncthreads();
  for (int off = 128; off; off >>= 1) {
    if (w < off) sm[w] = fmaxf(sm[w], sm[w + off]);
    __syncthreads();
  }
  float mx = sm[0];
  Mfull[w] = fmaxf(mean / mx, 0.5f);
}

// ---------------------------------------------------------------- attention pass B: full attention + masked residual
#define AST 72   // LDS row strides in halves (multiple of 8 -> 16B aligned frags)
__global__ __launch_bounds__(32)
void attn_apply_kernel(const h16* __restrict__ qkv, const float* __restrict__ rpe,
                       const float* __restrict__ Mfull, const float* __restrict__ resid,
                       h16* __restrict__ attOut, int idx) {
  __shared__ __align__(16) h16 shA[64 * AST];
  __shared__ __align__(16) h16 shV[32 * AST];

  const int lane = threadIdx.x;
  const int hs = lane >> 4, lr = lane & 15;
  const int w = blockIdx.x >> 4;
  const int h = blockIdx.x & 15;
  int start, sst;
  if (idx == 0) { start = w * 64; sst = 1; }
  else          { start = (w >> 6) * Lk + (w & 63); sst = 64; }
  const int ldq = sst * (3 * Ck);

  const h16* qb = qkv + (size_t)start * (3 * Ck) + h * HDk;
  const h16* kb = qkv + (size_t)start * (3 * Ck) + Ck + h * HDk;
  const h16* vb = qkv + (size_t)start * (3 * Ck) + 2 * Ck + h * HDk;
  const float* rp = rpe + h * (Sk * Sk);
  const float Mv = Mfull[w];

  // stage v^T into LDS: shV[d][t]
  for (int t = lane; t < 64; t += 32) {
    const h16* vp = vb + (size_t)t * ldq;
#pragma unroll
    for (int d = 0; d < HDk; ++d) shV[d * AST + t] = vp[d];
  }

  v16h kf[4];
#pragma unroll
  for (int nt = 0; nt < 4; ++nt)
    kf[nt] = load_frag(kb + (size_t)(nt * 16) * ldq, ldq, lane);

#pragma unroll
  for (int mt = 0; mt < 4; ++mt) {
    v16h qa = load_frag(qb + (size_t)(mt * 16) * ldq, ldq, lane);
    v8f c[4];
#pragma unroll
    for (int nt = 0; nt < 4; ++nt) { c[nt] = vzero8(); c[nt] = wmma16(qa, kf[nt], c[nt]); }
    // raw scores = qk + 0.1*rpe
#pragma unroll
    for (int nt = 0; nt < 4; ++nt)
#pragma unroll
      for (int r = 0; r < 8; ++r)
        c[nt][r] += 0.1f * rp[(mt * 16 + r + 8 * hs) * Sk + nt * 16 + lr];
    // row L2-normalize (row spans 16 lanes of this half x 4 n-tiles), trig, gate
#pragma unroll
    for (int r = 0; r < 8; ++r) {
      float ss = c[0][r]*c[0][r] + c[1][r]*c[1][r] + c[2][r]*c[2][r] + c[3][r]*c[3][r];
#pragma unroll
      for (int m = 8; m; m >>= 1) ss += __shfl_xor(ss, m, 32);  // masks<16: stays within half
      float inv = 1.0f / fmaxf(sqrtf(ss), 1e-12f);
      int srow = mt * 16 + r + 8 * hs;
#pragma unroll
      for (int nt = 0; nt < 4; ++nt) {
        float a = c[nt][r] * inv;
        float val = (1.0f - cosf(a * 1.57079632679489662f)) * Mv;
        shA[srow * AST + nt * 16 + lr] = (h16)val;
      }
    }
  }
  __syncthreads();  // single-wave block: S_NOP, LDS kept in-order per wave

  // out = attn @ v  (A from shA, B columns from shV rows)
  v16h vf[2][2];
#pragma unroll
  for (int ks = 0; ks < 2; ++ks)
#pragma unroll
    for (int n2 = 0; n2 < 2; ++n2)
      vf[ks][n2] = load_frag(shV + (n2 * 16) * AST + ks * 32, AST, lane);

#pragma unroll
  for (int mt = 0; mt < 4; ++mt) {
    v8f co[2] = { vzero8(), vzero8() };
#pragma unroll
    for (int ks = 0; ks < 2; ++ks) {
      v16h aa = load_frag(shA + (mt * 16) * AST + ks * 32, AST, lane);
#pragma unroll
      for (int n2 = 0; n2 < 2; ++n2) co[n2] = wmma16(aa, vf[ks][n2], co[n2]);
    }
#pragma unroll
    for (int n2 = 0; n2 < 2; ++n2)
#pragma unroll
      for (int r = 0; r < 8; ++r) {
        int s   = mt * 16 + r + 8 * hs;
        int tok = start + s * sst;
        int cc  = h * HDk + n2 * 16 + lr;
        size_t o = (size_t)tok * Ck + cc;
        float mres = 1.0f - Mfull[tok >> 6];   // residual chunking is row-major
        attOut[o] = (h16)(co[n2][r] + resid[o] * mres);
      }
  }
}

// ---------------------------------------------------------------- host
extern "C" void kernel_launch(void* const* d_in, const int* in_sizes, int n_in,
                              void* d_out, int out_size, void* d_ws, size_t ws_size,
                              hipStream_t stream) {
  (void)in_sizes; (void)n_in; (void)out_size; (void)ws_size;
  const float* x    = (const float*)d_in[0];
  const float* rpeA[2] = { (const float*)d_in[1], (const float*)d_in[2] };

  // workspace layout
  char* wsb = (char*)d_ws;
  size_t off = 0;
  auto take = [&](size_t bytes) { char* p = wsb + off; off += (bytes + 255) & ~(size_t)255; return p; };
  h16* wqkvT[2]; h16* wprojT[2]; h16* wfc1T[2]; h16* wfc2T[2];
  for (int b = 0; b < 2; ++b) {
    wqkvT[b] = (h16*)take((size_t)Ck * 3 * Ck * 2);
    wprojT[b] = (h16*)take((size_t)Ck * Ck * 2);
    wfc1T[b] = (h16*)take((size_t)Ck * 4 * Ck * 2);
    wfc2T[b] = (h16*)take((size_t)4 * Ck * Ck * 2);
  }
  h16* imgF = (h16*)take((size_t)Mrows * Ck * 2);
  // union region: [qkv | att] reused later by fc1 output g (exactly same size)
  char* uni = take((size_t)Mrows * 4 * Ck * 2);
  h16* qkvF = (h16*)uni;
  h16* attF = (h16*)(uni + (size_t)Mrows * 3 * Ck * 2);
  h16* gF   = (h16*)uni;
  float* Mpart = (float*)take(NWk * NHk * 4);
  float* Mfull = (float*)take(NWk * 4);
  float* xout = (float*)d_out;

  // convert + transpose all weights once
  for (int b = 0; b < 2; ++b) {
    int base = 3 + b * 11;
    const float* wqkv = (const float*)d_in[base + 2];
    const float* wproj = (const float*)d_in[base + 3];
    const float* wfc1 = (const float*)d_in[base + 7];
    const float* wfc2 = (const float*)d_in[base + 9];
    wconv_kernel<<<(Ck * 3 * Ck + 255) / 256, 256, 0, stream>>>(wqkv, wqkvT[b], Ck, 3 * Ck);
    wconv_kernel<<<(Ck * Ck + 255) / 256, 256, 0, stream>>>(wproj, wprojT[b], Ck, Ck);
    wconv_kernel<<<(Ck * 4 * Ck + 255) / 256, 256, 0, stream>>>(wfc1, wfc1T[b], Ck, 4 * Ck);
    wconv_kernel<<<(4 * Ck * Ck + 255) / 256, 256, 0, stream>>>(wfc2, wfc2T[b], 4 * Ck, Ck);
  }

  const dim3 blk256(256);
  for (int b = 0; b < 2; ++b) {
    int base = 3 + b * 11;
    const float* ln1g = (const float*)d_in[base + 0];
    const float* ln1b = (const float*)d_in[base + 1];
    const float* bproj = (const float*)d_in[base + 4];
    const float* ln2g = (const float*)d_in[base + 5];
    const float* ln2b = (const float*)d_in[base + 6];
    const float* bfc1 = (const float*)d_in[base + 8];
    const float* bfc2 = (const float*)d_in[base + 10];
    const float* streamIn = (b == 0) ? x : xout;   // residual stream at block input

    // LN1 -> img (f16)
    ln_kernel<<<Mrows / 8, blk256, 0, stream>>>(streamIn, ln1g, ln1b, imgF);
    // qkv = img @ wqkv  (f16 out)
    gemm_kernel<<<dim3(3 * Ck / 64, Mrows / 256), blk256, 0, stream>>>(
        imgF, wqkvT[b], nullptr, nullptr, nullptr, qkvF, 3 * Ck, Ck, 0);
    // attention gating stats -> M
    attn_mean_kernel<<<NWk * NHk, 32, 0, stream>>>(qkvF, rpeA[b], Mpart, b);
    mfinal_kernel<<<1, 256, 0, stream>>>(Mpart, Mfull);
    // full attention + masked residual -> attF (f16)
    attn_apply_kernel<<<NWk * NHk, 32, 0, stream>>>(qkvF, rpeA[b], Mfull, streamIn, attF, b);
    // x = stream + attF @ wproj + bproj  -> xout (f32)
    gemm_kernel<<<dim3(Ck / 64, Mrows / 256), blk256, 0, stream>>>(
        attF, wprojT[b], bproj, streamIn, xout, nullptr, Ck, Ck, 1);
    // LN2 -> img (f16)
    ln_kernel<<<Mrows / 8, blk256, 0, stream>>>(xout, ln2g, ln2b, imgF);
    // g = gelu(img @ wfc1 + bfc1)  (f16, aliases dead qkv/att region)
    gemm_kernel<<<dim3(4 * Ck / 64, Mrows / 256), blk256, 0, stream>>>(
        imgF, wfc1T[b], bfc1, nullptr, nullptr, gF, 4 * Ck, Ck, 2);
    // x = x + g @ wfc2 + bfc2  -> xout (f32, in-place same-thread RMW)
    gemm_kernel<<<dim3(Ck / 64, Mrows / 256), blk256, 0, stream>>>(
        gF, wfc2T[b], bfc2, xout, xout, nullptr, Ck, 4 * Ck, 1);
  }
}
